// VectorQuantizer_12927851561032
// MI455X (gfx1250) — matrix operand order repaired and checked by reading, compiled, not run on hardware
//
#include <hip/hip_runtime.h>

typedef __attribute__((ext_vector_type(16))) _Float16 v16h;
typedef __attribute__((ext_vector_type(8)))  float    v8f;

#define N_ROWS     262144
#define DIM        64
#define KCODES     1024
#define BLOCK_ROWS 128
#define NCHUNKS    (N_ROWS / BLOCK_ROWS)   // 2048 row-chunks total
#define PERSIST    512                     // persistent blocks
#define CHUNKS_PB  (NCHUNKS / PERSIST)     // 4 chunks per block
#define TILES      (KCODES / 16)           // 64 code tiles
#define LDS_STRIDE 72                      // halves per codebook row in LDS (64 + 8 pad)

__device__ __forceinline__ unsigned flip32(float f) {
  // order-preserving float -> unsigned (ashr, or-const, xor)
  unsigned u = __float_as_uint(f);
  return u ^ (unsigned)(((int)u >> 31) | (int)0x80000000);
}

// Pass 1: codebook f32 -> f16 copy + per-code squared norms (tiny: 256 KB read)
__global__ void vq_prep(const float* __restrict__ cb,
                        _Float16* __restrict__ cbH,
                        float* __restrict__ norms) {
  int k = blockIdx.x * blockDim.x + threadIdx.x;
  if (k < KCODES) {
    float acc = 0.f;
    for (int d = 0; d < DIM; ++d) {
      float v = cb[k * DIM + d];
      cbH[k * DIM + d] = (_Float16)v;
      acc += v * v;
    }
    norms[k] = acc;
  }
}

// Pass 2: distances via WMMA, argmin, gather, loss partials.
// Persistent: 512 blocks x 256 threads (8 waves); each wave owns 16 rows per chunk.
__global__ __launch_bounds__(256) void vq_main(
    const float* __restrict__ x, const float* __restrict__ cb,
    const _Float16* __restrict__ cbH, const float* __restrict__ norms,
    float* __restrict__ out, float* __restrict__ partials) {
  __shared__ _Float16 sB[KCODES * LDS_STRIDE];  // 144 KB, padded rows
  __shared__ float    sNorm[KCODES];            // 4 KB
  __shared__ float    sLoss[8];

  const int tid = threadIdx.x;

  // One-time stage of f16 codebook into LDS (padded stride), plus norms.
  {
    const uint2* src = (const uint2*)cbH;  // 16 x uint2 per 64-half row
    uint2*       dst = (uint2*)sB;         // 18 x uint2 per padded row
    for (int i = tid; i < KCODES * 16; i += 256) {
      int row = i >> 4, j = i & 15;
      dst[row * 18 + j] = src[i];
    }
    for (int i = tid; i < KCODES; i += 256) sNorm[i] = norms[i];
  }
  __syncthreads();

  const int w    = tid >> 5;
  const int lane = tid & 31;
  const int r    = lane & 15;
  const int hi   = lane >> 4;

  union BU { v16h v; uint4 q[2]; };

  auto ldTile = [&](int t, BU& b0, BU& b1, float& nr) {
    const int code = (t & (TILES - 1)) * 16 + r;
    const char* bp = (const char*)sB + code * (LDS_STRIDE * 2) + hi * 16;
    b0.q[0] = *(const uint4*)(bp);
    b0.q[1] = *(const uint4*)(bp + 32);
    b1.q[0] = *(const uint4*)(bp + 64);
    b1.q[1] = *(const uint4*)(bp + 96);
    nr = sNorm[code];
  };

  float lossTotal = 0.f;

  for (int chunk = 0; chunk < CHUNKS_PB; ++chunk) {
    const long chunkIdx = (long)blockIdx.x + (long)chunk * PERSIST;
    const long rowBase  = chunkIdx * BLOCK_ROWS + w * 16;

    // A-matrix (16x64 f32 -> two 16x32 f16 chunks, ISA 16-bit A layout):
    // lane<16 row M=r holds K {0..7, 16..23}; lane>=16 holds K {8..15, 24..31}.
    v16h a0, a1;
    {
      const float* xr = x + (rowBase + r) * DIM + hi * 8;
#pragma unroll
      for (int j = 0; j < 8; ++j) {
        a0[j]     = (_Float16)xr[j];
        a0[8 + j] = (_Float16)xr[16 + j];
        a1[j]     = (_Float16)xr[32 + j];
        a1[8 + j] = (_Float16)xr[48 + j];
      }
    }

    // Separate running minima: strict '<' with ascending code order keeps the
    // first (lowest) index on ties, matching jnp.argmin.
    float dmin[8];
    int   imin[8];
#pragma unroll
    for (int i = 0; i < 8; ++i) {
      dmin[i] = __uint_as_float(0x7F800000u);  // +inf
      imin[i] = 0;
    }

    auto step = [&](int t, const BU& b0, const BU& b1, float nr) {
      v8f c = {};
      c = __builtin_amdgcn_wmma_f32_16x16x32_f16(false, a0, false, b0.v, (short)0, c, false, false);
      c = __builtin_amdgcn_wmma_f32_16x16x32_f16(false, a1, false, b1.v, (short)0, c, false, false);
      const int code = t * 16 + r;
#pragma unroll
      for (int r8 = 0; r8 < 8; ++r8) {
        float dist = __builtin_fmaf(-2.0f, c[r8], nr);  // ||x||^2 dropped
        if (dist < dmin[r8]) {
          dmin[r8] = dist;
          imin[r8] = code;
        }
      }
    };

    // Software-pipelined sweep over 64 tiles: B loads issued one tile ahead.
    BU b0c, b1c, b0n, b1n;
    float nrC, nrN;
    ldTile(0, b0c, b1c, nrC);
    for (int t = 0; t < TILES; t += 2) {
      ldTile(t + 1, b0n, b1n, nrN);
      step(t, b0c, b1c, nrC);
      ldTile(t + 2, b0c, b1c, nrC);  // wraps harmlessly at t=62
      step(t + 1, b0n, b1n, nrN);
    }

    // Pack sortable 64-bit keys once, then xor-reduce across the 16 lanes
    // holding each row (halves independent). Equal dist bits -> lower code wins.
    unsigned long long best[8];
#pragma unroll
    for (int r8 = 0; r8 < 8; ++r8)
      best[r8] = ((unsigned long long)flip32(dmin[r8]) << 32) | (unsigned)imin[r8];
#pragma unroll
    for (int r8 = 0; r8 < 8; ++r8) {
#pragma unroll
      for (int m = 1; m < 16; m <<= 1) {
        unsigned long long o = __shfl_xor(best[r8], m, 32);
        if (o < best[r8]) best[r8] = o;
      }
    }

    // Gather exact fp32 codebook rows, write output, accumulate loss.
#pragma unroll
    for (int m = 0; m < 16; ++m) {
      int r8 = m & 7;
      unsigned long long kmin = __shfl(best[r8], (m < 8) ? 0 : 16, 32);
      int  idx = (int)(unsigned)kmin;
      long row = rowBase + m;
      float2 q  = *(const float2*)(cb + (long)idx * DIM + lane * 2);
      float2 xv = *(const float2*)(x + row * DIM + lane * 2);
      *(float2*)(out + row * DIM + lane * 2) = q;
      float dx = q.x - xv.x, dy = q.y - xv.y;
      lossTotal += dx * dx + dy * dy;
    }
  }

  // Per-block deterministic loss partial.
#pragma unroll
  for (int m = 1; m < 32; m <<= 1) lossTotal += __shfl_xor(lossTotal, m, 32);
  if (lane == 0) sLoss[w] = lossTotal;
  __syncthreads();
  if (tid == 0) {
    float s = 0.f;
#pragma unroll
    for (int i = 0; i < 8; ++i) s += sLoss[i];
    partials[blockIdx.x] = s;
  }
}

// Pass 3: deterministic reduction of block partials -> scalar loss.
__global__ void vq_finish(const float* __restrict__ partials, float* __restrict__ out) {
  __shared__ double sd[256];
  double a = 0.0;
  for (int i = threadIdx.x; i < PERSIST; i += 256) a += (double)partials[i];
  sd[threadIdx.x] = a;
  __syncthreads();
  for (int s = 128; s > 0; s >>= 1) {
    if ((int)threadIdx.x < s) sd[threadIdx.x] += sd[threadIdx.x + s];
    __syncthreads();
  }
  if (threadIdx.x == 0)
    out[(long)N_ROWS * DIM] = (float)(1.25 * sd[0] / ((double)N_ROWS * DIM));
}

extern "C" void kernel_launch(void* const* d_in, const int* in_sizes, int n_in,
                              void* d_out, int out_size, void* d_ws, size_t ws_size,
                              hipStream_t stream) {
  const float* x   = (const float*)d_in[0];  // [262144, 64] f32
  const float* cb  = (const float*)d_in[1];  // [1024, 64] f32
  float*       out = (float*)d_out;          // [262144*64] quantized + [1] loss

  // ws layout: f16 codebook (128 KB) | norms (4 KB) | block partials (2 KB)
  char*     ws       = (char*)d_ws;
  _Float16* cbH      = (_Float16*)ws;
  float*    norms    = (float*)(ws + (size_t)KCODES * DIM * sizeof(_Float16));
  float*    partials = (float*)(ws + (size_t)KCODES * DIM * sizeof(_Float16) +
                                KCODES * sizeof(float));

  vq_prep<<<(KCODES + 255) / 256, 256, 0, stream>>>(cb, cbH, norms);
  vq_main<<<PERSIST, 256, 0, stream>>>(x, cb, cbH, norms, out, partials);
  vq_finish<<<1, 256, 0, stream>>>(partials, out);
}
